// LSTMnetwork_18176301596946
// MI455X (gfx1250) — compile-verified, matrix operand order checked by reading, not run on hardware
//
#include <hip/hip_runtime.h>
#include <hip/hip_bf16.h>
#include <math.h>

// ---------------------------------------------------------------------------
// LSTM network on MI455X (gfx1250):
//   1) cvt f32->bf16 for x, W_ih, W_hh, W_out
//   2) WMMA bf16 GEMM: x_proj[S,4H] = x @ W_ih^T + (b_ih + b_hh)
//   3) persistent LDS-resident-weights recurrence (64 WGs, device barrier)
//   4) WMMA bf16 GEMM: logits[S,O] = hs @ W_out^T + b_out   (into d_out)
//   5) in-place log_softmax over rows of d_out
// ---------------------------------------------------------------------------

typedef __bf16 bf16;
typedef __attribute__((ext_vector_type(8)))  __bf16 v8bf;
typedef __attribute__((ext_vector_type(16))) __bf16 v16bf;
typedef __attribute__((ext_vector_type(8)))  float  v8f;

#define LSTM_S 8192
#define LSTM_I 1024
#define LSTM_H 1024
#define LSTM_O 1024
#define SCAN_NWG 64
#define SCAN_HB  16   // hidden units per workgroup (1024 / 64)

// --- optional gfx1250 async global->LDS path (ASYNCcnt-tracked) ------------
#if defined(__AMDGCN__) && \
    __has_builtin(__builtin_amdgcn_global_load_async_to_lds_b128) && \
    __has_builtin(__builtin_amdgcn_s_wait_asynccnt)
#define USE_ASYNC_LDS 1
#else
#define USE_ASYNC_LDS 0
#endif

#if USE_ASYNC_LDS
// Builtin signature (from hipcc diagnostic): arg0 is a pointer to a 16-byte
// int vector in the global (__device__) address space; arg1 is the LDS-side
// analogue; args 2/3 are immediate offset / cache-policy.
typedef int v4i32 __attribute__((vector_size(16)));
typedef __attribute__((address_space(1))) v4i32* gptr_b128;
typedef __attribute__((address_space(3))) v4i32* lptr_b128;

__device__ __forceinline__ void async_g2l_b128(const void* g, void* l) {
    __builtin_amdgcn_global_load_async_to_lds_b128((gptr_b128)g, (lptr_b128)l, 0, 0);
}
__device__ __forceinline__ void async_wait0() {
    __builtin_amdgcn_s_wait_asynccnt(0);
}
#endif

// ---------------------------------------------------------------------------
// f32 -> bf16 elementwise conversion
// ---------------------------------------------------------------------------
__global__ void cvt_f32_bf16(const float* __restrict__ in, bf16* __restrict__ out, int n) {
    int i = blockIdx.x * blockDim.x + threadIdx.x;
    if (i < n) out[i] = (bf16)in[i];
}

// ---------------------------------------------------------------------------
// zero h double-buffer + sync counters (ws is poisoned; must re-init per call)
// ---------------------------------------------------------------------------
__global__ void init_state(float* __restrict__ hbuf, unsigned* __restrict__ sync, int n) {
    int i = blockIdx.x * blockDim.x + threadIdx.x;
    if (i < n) hbuf[i] = 0.0f;
    if (i < 2) sync[i] = 0u;
}

// ---------------------------------------------------------------------------
// Fragment loader: 16 bf16 per lane as two global_load_b128.
// Per CDNA5 16-bit A layout: lanes 0-15 hold K{0..7,16..23}, lanes 16-31 hold
// K{8..15,24..31} of the 32-wide K step -> per-lane chunks at kk + h8*8 and
// kk + 16 + h8*8 within a contiguous K-major row.
// ---------------------------------------------------------------------------
__device__ __forceinline__ v16bf load_frag(const bf16* __restrict__ rowptr, int kk, int h8) {
    v8bf lo = *(const v8bf*)(rowptr + kk + h8 * 8);
    v8bf hi = *(const v8bf*)(rowptr + kk + 16 + h8 * 8);
    return __builtin_shufflevector(lo, hi, 0,1,2,3,4,5,6,7,8,9,10,11,12,13,14,15);
}

// ---------------------------------------------------------------------------
// WMMA bf16 GEMM:  C[M,N] = A[M,K] * B[N,K]^T + bias0 + bias1
// block = 256 threads = 8 waves; block tile 64(M) x 256(N);
// wave tile 32(M) x 64(N) = 2x4 v_wmma_f32_16x16x32_bf16 accumulators;
// B fragments reused across the two M-subtiles -> 12 b128 loads / 8 WMMAs.
// ---------------------------------------------------------------------------
__global__ void __launch_bounds__(256)
wmma_gemm_bias(const bf16* __restrict__ A, const bf16* __restrict__ B,
               float* __restrict__ C,
               const float* __restrict__ bias0, const float* __restrict__ bias1,
               int M, int N, int K) {
    const int tid  = threadIdx.x;
    const int wave = tid >> 5;
    const int lane = tid & 31;
    const int l16  = lane & 15;
    const int h8   = lane >> 4;

    const int m0 = blockIdx.y * 64  + (wave >> 2) * 32;
    const int n0 = blockIdx.x * 256 + (wave & 3) * 64;

    v8f acc0[4] = {};
    v8f acc1[4] = {};

    const bf16* __restrict__ arow0 = A + (size_t)(m0 + l16) * K;
    const bf16* __restrict__ arow1 = A + (size_t)(m0 + 16 + l16) * K;
    const bf16* __restrict__ brow  = B + (size_t)(n0 + l16) * K;

    for (int kk = 0; kk < K; kk += 32) {
        v16bf a0 = load_frag(arow0, kk, h8);
        v16bf a1 = load_frag(arow1, kk, h8);
#pragma unroll
        for (int nt = 0; nt < 4; ++nt) {
            v16bf bfr = load_frag(brow + (size_t)nt * 16 * K, kk, h8);
            acc0[nt] = __builtin_amdgcn_wmma_f32_16x16x32_bf16(
                false, a0, false, bfr, (short)0, acc0[nt], false, false);
            acc1[nt] = __builtin_amdgcn_wmma_f32_16x16x32_bf16(
                false, a1, false, bfr, (short)0, acc1[nt], false, false);
        }
    }

    // C/D layout: VGPR r holds M = r (lanes 0-15) / M = r+8 (lanes 16-31), N = lane&15
#pragma unroll
    for (int nt = 0; nt < 4; ++nt) {
        const int col = n0 + nt * 16 + l16;
        float bsum = 0.0f;
        if (bias0) bsum += bias0[col];
        if (bias1) bsum += bias1[col];
#pragma unroll
        for (int r = 0; r < 8; ++r) {
            const int row = m0 + r + h8 * 8;
            C[(size_t)row * N + col]        = acc0[nt][r] + bsum;
            C[(size_t)(row + 16) * N + col] = acc1[nt][r] + bsum;
        }
    }
}

// ---------------------------------------------------------------------------
// Persistent LSTM scan. 64 WGs x 256 threads. WG `b` owns hidden units
// [b*16, b*16+16). Its 64 W_hh rows (i,f,g,o gates x 16 units, 1024 wide each)
// live in LDS as bf16 (128 KB). Per step: stage h (4 KB, async->LDS when
// available), 4 lanes per gate-row do 256-MAC partial dots with ds_load_b128
// reads (packed bf16 decode + f32 fma), shfl-reduce, apply activations, write
// h_{t+1} to the alternate global buffer, then a device-wide sense barrier.
// Cell state c lives in registers of threads 0..15.
// ---------------------------------------------------------------------------
__global__ void __launch_bounds__(256)
lstm_scan(const bf16*  __restrict__ Whh,    // [4H, H] bf16
          const float* __restrict__ xproj,  // [S, 4H] f32 (bias already folded)
          float*       __restrict__ hbuf,   // [2, H] f32 double buffer
          bf16*        __restrict__ hs,     // [S, H] bf16 outputs
          unsigned*    __restrict__ sync)   // [0]=count, [1]=generation
{
    extern __shared__ unsigned char smem[];
    bf16*  wlds = (bf16*)smem;                                         // [64][H]
    float* hlds = (float*)(smem + (size_t)64 * LSTM_H * sizeof(bf16)); // [H]
    float* glds = hlds + LSTM_H;                                       // [64]

    const int tid   = threadIdx.x;
    const int wg    = blockIdx.x;
    const int hbase = wg * SCAN_HB;

    // Stage this WG's 64 weight rows into LDS: 8192 x 16B chunks.
    for (int c = tid; c < 64 * (LSTM_H / 8); c += blockDim.x) {
        const int r = c >> 7;           // weight row 0..63
        const int k = c & 127;          // 16B chunk within row
        const int g = r >> 4, j = r & 15;
        const bf16* src = Whh + (size_t)(g * LSTM_H + hbase + j) * LSTM_H + k * 8;
        bf16*       dst = wlds + (size_t)r * LSTM_H + k * 8;
#if USE_ASYNC_LDS
        async_g2l_b128(src, dst);
#else
        *(uint4*)dst = *(const uint4*)src;
#endif
    }
#if USE_ASYNC_LDS
    async_wait0();
#endif
    __syncthreads();

    const int row = tid >> 2;              // 0..63 : gate-row within WG
    const int sub = tid & 3;               // 0..3  : K-slice
    const int kpc = LSTM_H / 4;            // 256 elements per slice

    float c = 0.0f;                        // cell state (threads 0..15)

    for (int t = 0; t < LSTM_S; ++t) {
        // stage h_t into LDS (4 KB: one b128 per thread)
        const float* __restrict__ hsrc = hbuf + (size_t)(t & 1) * LSTM_H;
#if USE_ASYNC_LDS
        async_g2l_b128(hsrc + tid * 4, hlds + tid * 4);
        async_wait0();
#else
        *(float4*)(hlds + tid * 4) = *(const float4*)(hsrc + tid * 4);
#endif
        __builtin_prefetch(xproj + (size_t)(t + 1) * (4 * LSTM_H) + hbase, 0, 0);
        __syncthreads();

        // partial dot: 256 MACs via ds_load_b128 (8 bf16 weights / 4 f32 h)
        float acc = 0.0f;
        const uint4*  __restrict__ wq = (const uint4*)(wlds + (size_t)row * LSTM_H + sub * kpc);
        const float4* __restrict__ hq = (const float4*)(hlds + sub * kpc);
#pragma unroll 4
        for (int k = 0; k < kpc / 8; ++k) {
            const uint4 w = wq[k];
            const float4 ha = hq[2 * k];
            const float4 hb = hq[2 * k + 1];
            acc = fmaf(__uint_as_float(w.x << 16),          ha.x, acc);
            acc = fmaf(__uint_as_float(w.x & 0xffff0000u),  ha.y, acc);
            acc = fmaf(__uint_as_float(w.y << 16),          ha.z, acc);
            acc = fmaf(__uint_as_float(w.y & 0xffff0000u),  ha.w, acc);
            acc = fmaf(__uint_as_float(w.z << 16),          hb.x, acc);
            acc = fmaf(__uint_as_float(w.z & 0xffff0000u),  hb.y, acc);
            acc = fmaf(__uint_as_float(w.w << 16),          hb.z, acc);
            acc = fmaf(__uint_as_float(w.w & 0xffff0000u),  hb.w, acc);
        }
        acc += __shfl_xor(acc, 1, 32);
        acc += __shfl_xor(acc, 2, 32);
        if (sub == 0) {
            const int g = row >> 4, j = row & 15;
            glds[row] = acc + xproj[(size_t)t * (4 * LSTM_H) + g * LSTM_H + hbase + j];
        }
        __syncthreads();

        if (tid < SCAN_HB) {
            const int j = tid;
            const float gi = glds[ 0 + j];
            const float gf = glds[16 + j];
            const float gg = glds[32 + j];
            const float go = glds[48 + j];
            const float si = 1.0f / (1.0f + __expf(-gi));
            const float sf = 1.0f / (1.0f + __expf(-gf));
            const float tg = tanhf(gg);
            const float so = 1.0f / (1.0f + __expf(-go));
            c = sf * c + si * tg;
            const float h = so * tanhf(c);
            hbuf[(size_t)((t + 1) & 1) * LSTM_H + hbase + j] = h;
            hs[(size_t)t * LSTM_H + hbase + j] = (bf16)h;
        }
        __syncthreads();

        // device-wide barrier (sense = generation counter == t+1)
        if (tid == 0) {
            __threadfence();
            const unsigned arrived = atomicAdd(&sync[0], 1u);
            if (arrived == SCAN_NWG - 1) {
                sync[0] = 0;
                __threadfence();
                atomicAdd(&sync[1], 1u);
            } else {
                while (__hip_atomic_load(&sync[1], __ATOMIC_RELAXED,
                                         __HIP_MEMORY_SCOPE_AGENT) < (unsigned)(t + 1)) {
                    __builtin_amdgcn_s_sleep(2);
                }
            }
        }
        __syncthreads();
    }
}

// ---------------------------------------------------------------------------
// In-place row-wise log_softmax: out[r, :] -= (max + log(sum(exp(x - max))))
// ---------------------------------------------------------------------------
__global__ void __launch_bounds__(256)
log_softmax_rows(float* __restrict__ out, int N) {
    __shared__ float red[256];
    float* __restrict__ p = out + (size_t)blockIdx.x * N;
    const int tid = threadIdx.x;

    float m = -INFINITY;
    for (int i = tid; i < N; i += 256) m = fmaxf(m, p[i]);
    red[tid] = m; __syncthreads();
    for (int s = 128; s > 0; s >>= 1) {
        if (tid < s) red[tid] = fmaxf(red[tid], red[tid + s]);
        __syncthreads();
    }
    m = red[0]; __syncthreads();

    float sum = 0.0f;
    for (int i = tid; i < N; i += 256) sum += __expf(p[i] - m);
    red[tid] = sum; __syncthreads();
    for (int s = 128; s > 0; s >>= 1) {
        if (tid < s) red[tid] += red[tid + s];
        __syncthreads();
    }
    const float lse = m + __logf(red[0]);
    __syncthreads();

    for (int i = tid; i < N; i += 256) p[i] = p[i] - lse;
}

// ---------------------------------------------------------------------------
// Host launcher
// ---------------------------------------------------------------------------
extern "C" void kernel_launch(void* const* d_in, const int* in_sizes, int n_in,
                              void* d_out, int out_size, void* d_ws, size_t ws_size,
                              hipStream_t stream) {
    (void)in_sizes; (void)n_in; (void)out_size; (void)ws_size;

    const float* x     = (const float*)d_in[0];
    const float* W_ih  = (const float*)d_in[1];
    const float* W_hh  = (const float*)d_in[2];
    const float* b_ih  = (const float*)d_in[3];
    const float* b_hh  = (const float*)d_in[4];
    const float* W_out = (const float*)d_in[5];
    const float* b_out = (const float*)d_in[6];

    constexpr int S = LSTM_S, I = LSTM_I, H = LSTM_H, O = LSTM_O;
    constexpr int G = 4 * H;

    // workspace carve-up
    char* ws = (char*)d_ws;
    size_t off = 0;
    auto carve = [&](size_t bytes) -> void* {
        void* p = ws + off;
        off = (off + bytes + 255) & ~(size_t)255;
        return p;
    };
    bf16*     xbf    = (bf16*)    carve((size_t)S * I * sizeof(bf16));
    bf16*     wihbf  = (bf16*)    carve((size_t)G * I * sizeof(bf16));
    bf16*     whhbf  = (bf16*)    carve((size_t)G * H * sizeof(bf16));
    bf16*     woutbf = (bf16*)    carve((size_t)O * H * sizeof(bf16));
    bf16*     hsbf   = (bf16*)    carve((size_t)S * H * sizeof(bf16));
    float*    xproj  = (float*)   carve((size_t)S * G * sizeof(float));
    float*    hbuf   = (float*)   carve((size_t)2 * H * sizeof(float));
    unsigned* sync   = (unsigned*)carve(256);

    const int cvtB = 256;
    auto nblk = [](int n, int b) { return (n + b - 1) / b; };

    // 0) init persistent state
    init_state<<<nblk(2 * H, cvtB), cvtB, 0, stream>>>(hbuf, sync, 2 * H);

    // 1) conversions f32 -> bf16
    cvt_f32_bf16<<<nblk(S * I, cvtB), cvtB, 0, stream>>>(x,     xbf,    S * I);
    cvt_f32_bf16<<<nblk(G * I, cvtB), cvtB, 0, stream>>>(W_ih,  wihbf,  G * I);
    cvt_f32_bf16<<<nblk(G * H, cvtB), cvtB, 0, stream>>>(W_hh,  whhbf,  G * H);
    cvt_f32_bf16<<<nblk(O * H, cvtB), cvtB, 0, stream>>>(W_out, woutbf, O * H);

    // 2) x_proj = x @ W_ih^T + (b_ih + b_hh)   [S, 4H]
    {
        dim3 grid(G / 256, S / 64);
        wmma_gemm_bias<<<grid, 256, 0, stream>>>(xbf, wihbf, xproj,
                                                 b_ih, b_hh, S, G, I);
    }

    // 3) recurrence (persistent, LDS-resident W_hh)
    {
        const size_t smem = (size_t)64 * H * sizeof(bf16)  // weights
                          + (size_t)H * sizeof(float)      // h staging
                          + (size_t)64 * sizeof(float);    // gate scratch
        lstm_scan<<<SCAN_NWG, 256, smem, stream>>>(whhbf, xproj, hbuf, hsbf, sync);
    }

    // 4) logits = hs @ W_out^T + b_out  -> d_out
    {
        dim3 grid(O / 256, S / 64);
        wmma_gemm_bias<<<grid, 256, 0, stream>>>(hsbf, woutbf, (float*)d_out,
                                                 b_out, nullptr, S, O, H);
    }

    // 5) log_softmax in place
    log_softmax_rows<<<S, 256, 0, stream>>>((float*)d_out, O);
}
